// CustomMultiheadAttention_43877385896502
// MI455X (gfx1250) — compile-verified, matrix operand order checked by reading, not run on hardware
//
#include <hip/hip_runtime.h>
#include <hip/hip_bf16.h>

// ---------------------------------------------------------------------------
// CDNA5 (gfx1250) multi-head attention forward with RoPE.
// All GEMM stages use v_wmma_f32_16x16x32_f16 (f16 inputs, f32 accumulate).
// ---------------------------------------------------------------------------

typedef _Float16 v8h  __attribute__((ext_vector_type(8)));
typedef _Float16 v16h __attribute__((ext_vector_type(16)));
typedef float    v8f  __attribute__((ext_vector_type(8)));

#define EMBED   1024
#define HEADS   16
#define HDIM    64
#define BATCH   2
#define SEQ     2048
#define BH      (BATCH * HEADS)
#define MROWS   (BATCH * SEQ)        // 4096 rows for the projection GEMMs

static __device__ __forceinline__ v16h hcat(v8h lo, v8h hi) {
  return __builtin_shufflevector(lo, hi, 0,1,2,3,4,5,6,7,8,9,10,11,12,13,14,15);
}

static __device__ __forceinline__ v8f wmma16(v16h a, v16h b, v8f c) {
  // D = A(16x32 f16) * B(32x16 f16) + C(16x16 f32)
  return __builtin_amdgcn_wmma_f32_16x16x32_f16(
      /*neg_a=*/false, a, /*neg_b=*/false, b,
      /*c_mod=*/(short)0, c, /*reuse_a=*/false, /*reuse_b=*/false);
}

// "A pattern" 16x32 f16 fragment from row-major storage (also valid for the
// B operand of C = A*B^T since both operands then load identically).
// Lane l: row = l&15; K elems [8*(l>>4) .. +7] and [16+8*(l>>4) .. +7].
static __device__ __forceinline__ v16h frag_h(const _Float16* tile, int stride, int lane) {
  const int r  = lane & 15;
  const int ks = (lane >> 4) << 3;
  const _Float16* p = tile + (size_t)r * (size_t)stride + ks;
  v8h lo = *(const v8h*)(p);
  v8h hi = *(const v8h*)(p + 16);
  return hcat(lo, hi);
}

// Same fragment but sourced from f32 (converted in-register) — used to feed
// the softmax probabilities (stored f32 in d_out) straight into WMMA.
static __device__ __forceinline__ v16h frag_f32(const float* tile, size_t stride, int lane) {
  const int r  = lane & 15;
  const int ks = (lane >> 4) << 3;
  const float* p = tile + (size_t)r * stride + ks;
  float4 a0 = *(const float4*)(p);
  float4 a1 = *(const float4*)(p + 4);
  float4 b0 = *(const float4*)(p + 16);
  float4 b1 = *(const float4*)(p + 20);
  v16h v;
  v[0]=(_Float16)a0.x;  v[1]=(_Float16)a0.y;  v[2]=(_Float16)a0.z;  v[3]=(_Float16)a0.w;
  v[4]=(_Float16)a1.x;  v[5]=(_Float16)a1.y;  v[6]=(_Float16)a1.z;  v[7]=(_Float16)a1.w;
  v[8]=(_Float16)b0.x;  v[9]=(_Float16)b0.y;  v[10]=(_Float16)b0.z; v[11]=(_Float16)b0.w;
  v[12]=(_Float16)b1.x; v[13]=(_Float16)b1.y; v[14]=(_Float16)b1.z; v[15]=(_Float16)b1.w;
  return v;
}

static __device__ __forceinline__ void cvt_store8(_Float16* dst, float4 a, float4 b) {
  v8h v;
  v[0]=(_Float16)a.x; v[1]=(_Float16)a.y; v[2]=(_Float16)a.z; v[3]=(_Float16)a.w;
  v[4]=(_Float16)b.x; v[5]=(_Float16)b.y; v[6]=(_Float16)b.z; v[7]=(_Float16)b.w;
  *(v8h*)dst = v;
}

// ---------------------------------------------------------------------------
// GEMM: Y[M,N] = A[M,K] * W[N,K]^T + bias[N]   (all f32 in global; f16 WMMA)
// 128x128 tile per 256-thread WG (8 waves, each 64(m) x 32(n)); K-step 32.
// f32 -> f16 conversion happens during the cooperative LDS fill.
// ---------------------------------------------------------------------------
#define GTM 128
#define GTN 128
#define GTK 32
#define LDT 40   // padded LDS row stride in halfs (80B -> conflict-free frag reads)

__global__ __launch_bounds__(256) void gemm_bias_kernel(
    const float* __restrict__ A, const float* __restrict__ W,
    const float* __restrict__ bias, float* __restrict__ Y,
    int M, int N, int K)
{
  __shared__ _Float16 At[GTM * LDT];
  __shared__ _Float16 Wt[GTN * LDT];

  const int t    = threadIdx.x;
  const int lane = t & 31;
  const int wv   = t >> 5;
  const int bm   = (blockIdx.x % (M / GTM)) * GTM;
  const int bn   = (blockIdx.x / (M / GTM)) * GTN;
  const int wm   = (wv & 1) * 64;   // 2 waves in m
  const int wn   = (wv >> 1) * 32;  // 4 waves in n

  v8f acc[4][2];
#pragma unroll
  for (int i = 0; i < 4; ++i)
#pragma unroll
    for (int j = 0; j < 2; ++j) acc[i][j] = (v8f)0.0f;

  const int frow = t >> 1;
  const int fcb  = (t & 1) * 16;

  for (int k0 = 0; k0 < K; k0 += GTK) {
    {
      const float* ap = A + (size_t)(bm + frow) * K + k0 + fcb;
      float4 a0 = *(const float4*)(ap);
      float4 a1 = *(const float4*)(ap + 4);
      float4 a2 = *(const float4*)(ap + 8);
      float4 a3 = *(const float4*)(ap + 12);
      cvt_store8(At + frow * LDT + fcb,     a0, a1);
      cvt_store8(At + frow * LDT + fcb + 8, a2, a3);
      const float* wp = W + (size_t)(bn + frow) * K + k0 + fcb;
      float4 w0 = *(const float4*)(wp);
      float4 w1 = *(const float4*)(wp + 4);
      float4 w2 = *(const float4*)(wp + 8);
      float4 w3 = *(const float4*)(wp + 12);
      cvt_store8(Wt + frow * LDT + fcb,     w0, w1);
      cvt_store8(Wt + frow * LDT + fcb + 8, w2, w3);
    }
    __syncthreads();

    v16h af[4], bf[2];
#pragma unroll
    for (int i = 0; i < 4; ++i) af[i] = frag_h(At + (wm + i * 16) * LDT, LDT, lane);
#pragma unroll
    for (int j = 0; j < 2; ++j) bf[j] = frag_h(Wt + (wn + j * 16) * LDT, LDT, lane);
#pragma unroll
    for (int i = 0; i < 4; ++i)
#pragma unroll
      for (int j = 0; j < 2; ++j) acc[i][j] = wmma16(af[i], bf[j], acc[i][j]);
    __syncthreads();
  }

  const int col = lane & 15;
  const int mhi = (lane >> 4) * 8;
#pragma unroll
  for (int j = 0; j < 2; ++j) {
    const int n = bn + wn + j * 16 + col;
    const float bv = bias[n];
#pragma unroll
    for (int i = 0; i < 4; ++i) {
      const int m0 = bm + wm + i * 16 + mhi;
#pragma unroll
      for (int r = 0; r < 8; ++r)
        Y[(size_t)(m0 + r) * N + n] = acc[i][j][r] + bv;
    }
  }
}

// ---------------------------------------------------------------------------
// RoPE + repack: f32 [B,S,E] -> f16 [B,H,S,D] with rotary embedding applied.
// One thread per (b,s,h,pair); pair = HDIM/2 = 32.
// ---------------------------------------------------------------------------
__global__ __launch_bounds__(256) void rope_pack_kernel(
    const float* __restrict__ X, _Float16* __restrict__ Y)
{
  const int idx  = blockIdx.x * 256 + threadIdx.x;   // 21 bits total
  const int pair = idx & 31;
  const int h    = (idx >> 5) & 15;
  const int s    = (idx >> 9) & 2047;
  const int b    = idx >> 20;

  const float* src = X + ((size_t)(b * SEQ + s)) * EMBED + h * HDIM + pair * 2;
  const float xr = src[0], xi = src[1];
  const float inv = powf(10000.0f, -(2.0f * (float)pair) / (float)HDIM);
  const float ang = (float)s * inv;
  const float c = cosf(ang), sn = sinf(ang);

  _Float16* dst = Y + (((size_t)(b * HEADS + h)) * SEQ + s) * HDIM + pair * 2;
  dst[0] = (_Float16)(xr * c - xi * sn);
  dst[1] = (_Float16)(xr * sn + xi * c);
}

// V repack: f32 [B,S,E] -> f16 [B,H,D,S] (transposed so the PV B-operand
// fragment loads are contiguous B128s).
__global__ __launch_bounds__(256) void vtrans_pack_kernel(
    const float* __restrict__ X, _Float16* __restrict__ Y)
{
  const int idx = blockIdx.x * 256 + threadIdx.x;    // 22 bits total
  const int d = idx & 63;
  const int h = (idx >> 6) & 15;
  const int s = (idx >> 10) & 2047;
  const int b = idx >> 21;
  const float v = X[((size_t)(b * SEQ + s)) * EMBED + h * HDIM + d];
  Y[(((size_t)(b * HEADS + h)) * HDIM + d) * SEQ + s] = (_Float16)v;
}

// ---------------------------------------------------------------------------
// Scores: attn[b,h,q,k] = (Q . K) / sqrt(D).  64x64 tile per wave, D=64.
// Both operands row-major [B,H,S,D] -> identical "A pattern" frag loads.
// ---------------------------------------------------------------------------
__global__ __launch_bounds__(256) void scores_kernel(
    const _Float16* __restrict__ Qh, const _Float16* __restrict__ Kh,
    float* __restrict__ attn)
{
  const int lane = threadIdx.x & 31;
  const int wid  = blockIdx.x * 8 + (threadIdx.x >> 5);
  const int QT = SEQ / 64, KT = SEQ / 64;
  const int bh  = wid / (QT * KT);
  const int rem = wid % (QT * KT);
  const int qt  = rem / KT, kt = rem % KT;

  const _Float16* Qb = Qh + (size_t)bh * SEQ * HDIM + (size_t)qt * 64 * HDIM;
  const _Float16* Kb = Kh + (size_t)bh * SEQ * HDIM + (size_t)kt * 64 * HDIM;

  v8f acc[4][4];
#pragma unroll
  for (int i = 0; i < 4; ++i)
#pragma unroll
    for (int j = 0; j < 4; ++j) acc[i][j] = (v8f)0.0f;

#pragma unroll
  for (int kk = 0; kk < HDIM; kk += 32) {
    v16h a[4], b[4];
#pragma unroll
    for (int i = 0; i < 4; ++i) a[i] = frag_h(Qb + (size_t)(i * 16) * HDIM + kk, HDIM, lane);
#pragma unroll
    for (int j = 0; j < 4; ++j) b[j] = frag_h(Kb + (size_t)(j * 16) * HDIM + kk, HDIM, lane);
#pragma unroll
    for (int i = 0; i < 4; ++i)
#pragma unroll
      for (int j = 0; j < 4; ++j) acc[i][j] = wmma16(a[i], b[j], acc[i][j]);
  }

  const int col = lane & 15;
  const int mhi = (lane >> 4) * 8;
  const float scale = 0.125f;  // 1/sqrt(64)
#pragma unroll
  for (int i = 0; i < 4; ++i)
#pragma unroll
    for (int j = 0; j < 4; ++j) {
      float* op = attn + ((size_t)bh * SEQ + qt * 64 + i * 16 + mhi) * SEQ
                       + (size_t)kt * 64 + j * 16 + col;
#pragma unroll
      for (int r = 0; r < 8; ++r) op[(size_t)r * SEQ] = acc[i][j][r] * scale;
    }
}

// ---------------------------------------------------------------------------
// Row softmax in place over the 2048-wide attn rows. One WG per row.
// ---------------------------------------------------------------------------
__global__ __launch_bounds__(256) void softmax_kernel(float* __restrict__ attn)
{
  __shared__ float red[256];
  float* p = attn + (size_t)blockIdx.x * SEQ;
  const int t = threadIdx.x;

  float vals[8];
  float m = -3.0e38f;
#pragma unroll
  for (int i = 0; i < 8; ++i) { vals[i] = p[t + i * 256]; m = fmaxf(m, vals[i]); }
  red[t] = m; __syncthreads();
  for (int s = 128; s > 0; s >>= 1) {
    if (t < s) red[t] = fmaxf(red[t], red[t + s]);
    __syncthreads();
  }
  m = red[0]; __syncthreads();

  float sum = 0.0f;
#pragma unroll
  for (int i = 0; i < 8; ++i) { vals[i] = __expf(vals[i] - m); sum += vals[i]; }
  red[t] = sum; __syncthreads();
  for (int s = 128; s > 0; s >>= 1) {
    if (t < s) red[t] += red[t + s];
    __syncthreads();
  }
  const float inv = 1.0f / red[0];
#pragma unroll
  for (int i = 0; i < 8; ++i) p[t + i * 256] = vals[i] * inv;
}

// ---------------------------------------------------------------------------
// PV: ctx[b,q,h*64+d] = sum_k P[b,h,q,k] * V[b,h,k,d].  32(q) x 64(d) per
// wave; P read f32 (converted in-register), V from transposed [B,H,D,S].
// ---------------------------------------------------------------------------
__global__ __launch_bounds__(256) void pv_kernel(
    const float* __restrict__ attn, const _Float16* __restrict__ Vt,
    float* __restrict__ ctx)
{
  const int lane = threadIdx.x & 31;
  const int wid  = blockIdx.x * 8 + (threadIdx.x >> 5);
  const int QT = SEQ / 32;  // 64
  const int bh = wid / QT, qt = wid % QT;
  const int b = bh / HEADS, h = bh % HEADS;

  const float*    Pb = attn + ((size_t)bh * SEQ + (size_t)qt * 32) * SEQ;
  const _Float16* Vb = Vt + (size_t)bh * HDIM * SEQ;

  v8f acc[2][4];
#pragma unroll
  for (int i = 0; i < 2; ++i)
#pragma unroll
    for (int j = 0; j < 4; ++j) acc[i][j] = (v8f)0.0f;

  for (int k0 = 0; k0 < SEQ; k0 += 32) {
    v16h a[2], bf[4];
#pragma unroll
    for (int i = 0; i < 2; ++i) a[i] = frag_f32(Pb + (size_t)(i * 16) * SEQ + k0, SEQ, lane);
#pragma unroll
    for (int j = 0; j < 4; ++j) bf[j] = frag_h(Vb + (size_t)(j * 16) * SEQ + k0, SEQ, lane);
#pragma unroll
    for (int i = 0; i < 2; ++i)
#pragma unroll
      for (int j = 0; j < 4; ++j) acc[i][j] = wmma16(a[i], bf[j], acc[i][j]);
  }

  const int col = lane & 15;
  const int mhi = (lane >> 4) * 8;
#pragma unroll
  for (int i = 0; i < 2; ++i)
#pragma unroll
    for (int j = 0; j < 4; ++j) {
      const int q0 = qt * 32 + i * 16 + mhi;
      const int d0 = h * HDIM + j * 16 + col;
      float* op = ctx + ((size_t)b * SEQ + q0) * EMBED + d0;
#pragma unroll
      for (int r = 0; r < 8; ++r) op[(size_t)r * EMBED] = acc[i][j][r];
    }
}

// ---------------------------------------------------------------------------
// Host-side orchestration (graph-capture safe: launches only).
// ---------------------------------------------------------------------------
extern "C" void kernel_launch(void* const* d_in, const int* in_sizes, int n_in,
                              void* d_out, int out_size, void* d_ws, size_t ws_size,
                              hipStream_t stream) {
  (void)in_sizes; (void)n_in; (void)out_size; (void)ws_size;
  const float* query = (const float*)d_in[0];
  const float* key   = (const float*)d_in[1];
  const float* value = (const float*)d_in[2];
  const float* Wq = (const float*)d_in[3];
  const float* bq = (const float*)d_in[4];
  const float* Wk = (const float*)d_in[5];
  const float* bk = (const float*)d_in[6];
  const float* Wv = (const float*)d_in[7];
  const float* bv = (const float*)d_in[8];
  const float* Wo = (const float*)d_in[9];
  const float* bo = (const float*)d_in[10];

  float* out  = (float*)d_out;                              // [B,S,E]
  float* attn = out + (size_t)BATCH * SEQ * EMBED;          // [B,H,S,S]

  // Workspace layout (~72 MB):
  //  [0, szP)        Qp f32   (aliased later by ctx — Qp dead after rope_pack)
  //  [szP, 2szP)     Kp f32
  //  [2szP, 3szP)    Vp f32
  //  then Qh, Kh f16 [B,H,S,D] and Vt f16 [B,H,D,S]
  char* ws = (char*)d_ws;
  const size_t elems = (size_t)BATCH * SEQ * EMBED;         // 4,194,304
  const size_t szP = elems * sizeof(float);                 // 16 MB
  float* Qp = (float*)(ws);
  float* Kp = (float*)(ws + szP);
  float* Vp = (float*)(ws + 2 * szP);
  _Float16* Qh = (_Float16*)(ws + 3 * szP);
  _Float16* Kh = Qh + elems;
  _Float16* Vt = Kh + elems;
  float* ctx = Qp;  // alias: Qp no longer needed once Qh exists

  const dim3 blk(256);
  const int gemm_grid = (MROWS / GTM) * (EMBED / GTN);      // 32 * 8 = 256

  // 1) Projections (f32 out, bias fused)
  gemm_bias_kernel<<<gemm_grid, blk, 0, stream>>>(query, Wq, bq, Qp, MROWS, EMBED, EMBED);
  gemm_bias_kernel<<<gemm_grid, blk, 0, stream>>>(key,   Wk, bk, Kp, MROWS, EMBED, EMBED);
  gemm_bias_kernel<<<gemm_grid, blk, 0, stream>>>(value, Wv, bv, Vp, MROWS, EMBED, EMBED);

  // 2) RoPE + repack to head-major f16; V to transposed layout
  rope_pack_kernel<<<(BATCH * SEQ * HEADS * (HDIM / 2)) / 256, blk, 0, stream>>>(Qp, Qh);
  rope_pack_kernel<<<(BATCH * SEQ * HEADS * (HDIM / 2)) / 256, blk, 0, stream>>>(Kp, Kh);
  vtrans_pack_kernel<<<(int)(elems / 256), blk, 0, stream>>>(Vp, Vt);

  // 3) Raw scores -> d_out attn region (scaled)
  scores_kernel<<<(BH * (SEQ / 64) * (SEQ / 64)) / 8, blk, 0, stream>>>(Qh, Kh, attn);

  // 4) Softmax in place (these ARE the returned attn_weights)
  softmax_kernel<<<BH * SEQ, blk, 0, stream>>>(attn);

  // 5) context = P @ V
  pv_kernel<<<(BH * (SEQ / 32)) / 8, blk, 0, stream>>>(attn, Vt, ctx);

  // 6) attn_output = ctx @ Wo^T + bo
  gemm_bias_kernel<<<gemm_grid, blk, 0, stream>>>(ctx, Wo, bo, out, MROWS, EMBED, EMBED);
}
